// Attention_1778116460685
// MI455X (gfx1250) — matrix operand order repaired and checked by reading, not compiled, not run on hardware
//
#include <hip/hip_runtime.h>

typedef __attribute__((ext_vector_type(16))) _Float16 v16h;
typedef __attribute__((ext_vector_type(8)))  _Float16 v8h;
typedef __attribute__((ext_vector_type(4)))  _Float16 v4h;
typedef __attribute__((ext_vector_type(2)))  _Float16 h2;
typedef __attribute__((ext_vector_type(8)))  float    v8f;
typedef __attribute__((ext_vector_type(8)))  unsigned v8u;

#define ATT_S 2048
#define ATT_H 16
#define ATT_D 128
#define QK_SCALE 0.08838834764831845f                       // 1/sqrt(128)
#define QK_SCALE_LOG2 (0.08838834764831845f * 1.4426950408889634f)  // * log2(e)

// D = A*B + C, fp16 inputs, fp32 accumulate
__device__ __forceinline__ v8f wmma_f16(v16h a, v16h b, v8f c) {
    return __builtin_amdgcn_wmma_f32_16x16x32_f16(
        false, a, false, b, (short)0, c, false, false);
}

// xor-16 half-swap via v_permlanex16_b32 (VALU; avoids ds_bpermute+dscnt wait).
// Identity nibble selects: lane i reads lane i of the opposite 16-lane row.
__device__ __forceinline__ unsigned permx16u(unsigned v) {
    return (unsigned)__builtin_amdgcn_permlanex16(
        (int)v, (int)v, 0x76543210, 0xfedcba98, false, false);
}
__device__ __forceinline__ float permx16f(float v) {
    return __builtin_bit_cast(float, (unsigned)__builtin_amdgcn_permlanex16(
        __builtin_bit_cast(int, v), __builtin_bit_cast(int, v),
        0x76543210, 0xfedcba98, false, false));
}

// WGP-scope prefetch (pulls into all cache levels; __builtin_prefetch can
// only reach SE/DEV scope which skips the WGP caches).
__device__ __forceinline__ void prefetch_wgp(const void* p) {
    asm volatile("global_prefetch_b8 %0, off" :: "v"(p));
}

// =====================  pre-conversion kernels  =====================

// Qh[i] = f16(Q[i]*scale*log2e); Kh[i] = f16(K[i])   (4 elements / thread)
__global__ __launch_bounds__(256)
void cvt_qk(const float* __restrict__ Q, const float* __restrict__ K,
            _Float16* __restrict__ Qh, _Float16* __restrict__ Kh) {
    size_t i = ((size_t)blockIdx.x * 256 + threadIdx.x) * 4;
    float4 q = *(const float4*)(Q + i);
    float4 k = *(const float4*)(K + i);
    v4h qh = { (_Float16)(q.x * QK_SCALE_LOG2), (_Float16)(q.y * QK_SCALE_LOG2),
               (_Float16)(q.z * QK_SCALE_LOG2), (_Float16)(q.w * QK_SCALE_LOG2) };
    v4h kh = { (_Float16)k.x, (_Float16)k.y, (_Float16)k.z, (_Float16)k.w };
    *(v4h*)(Qh + i) = qh;
    *(v4h*)(Kh + i) = kh;
}

// Vt[h][d][s] = f16(V[s][h][d])   — 32x32 LDS tile transpose per block
__global__ __launch_bounds__(256)
void cvt_v_t(const float* __restrict__ V, _Float16* __restrict__ Vt) {
    __shared__ float tile[32][33];
    const int sb = blockIdx.x * 32, db = blockIdx.y * 32, h = blockIdx.z;
    const int tx = threadIdx.x & 31, ty = threadIdx.x >> 5;   // ty: 0..7
#pragma unroll
    for (int i = 0; i < 4; ++i) {
        int s = ty + i * 8;
        tile[s][tx] = V[((size_t)(sb + s) * ATT_H + h) * ATT_D + db + tx];
    }
    __syncthreads();
#pragma unroll
    for (int i = 0; i < 4; ++i) {
        int d = ty + i * 8;
        Vt[((size_t)h * ATT_D + db + d) * ATT_S + sb + tx] = (_Float16)tile[tx][d];
    }
}

// =====================  fp16 flash-attention kernel  =====================
// Transposed matmuls: S^T = K·Q^T, O^T = V^T·P^T. Scores are in log2 domain
// (log2e folded into Qh), so softmax uses native exp2.
// Per wave: one head, 16 query rows. nm = query (B/C col) or d/key (A row).

template<bool MASKED>
__device__ __forceinline__ void fa_block(
    const _Float16* __restrict__ Kh, const _Float16* __restrict__ vhead,
    int h, int kb, int qb, int nm, int half,
    const v16h (&qf)[4], v8f (&acc)[8], float& m, float& l)
{
    const _Float16* krow0 = Kh + ((size_t)(kb + nm) * ATT_H + h) * ATT_D;
    const _Float16* krow1 = krow0 + (size_t)16 * ATT_H * ATT_D;

    if (!MASKED) {   // prefetch next key block into WGP caches
        prefetch_wgp(krow0 + (size_t)32 * ATT_H * ATT_D);
        prefetch_wgp(vhead + kb + 32);
    }

    // ---- S^T = K·Q^T : two 16(key)x16(query) tiles ----
    v8f s0 = {}, s1 = {};
#pragma unroll
    for (int c = 0; c < 4; ++c) {
        v8h a0 = *(const v8h*)(krow0 + c * 32 + half * 16);
        v8h a1 = *(const v8h*)(krow0 + c * 32 + half * 16 + 8);
        v8h b0 = *(const v8h*)(krow1 + c * 32 + half * 16);
        v8h b1 = *(const v8h*)(krow1 + c * 32 + half * 16 + 8);
        v16h ka = __builtin_shufflevector(a0, a1, 0,1,2,3,4,5,6,7,8,9,10,11,12,13,14,15);
        v16h kc = __builtin_shufflevector(b0, b1, 0,1,2,3,4,5,6,7,8,9,10,11,12,13,14,15);
        s0 = wmma_f16(ka, qf[c], s0);
        s1 = wmma_f16(kc, qf[c], s1);
    }

    // ---- online softmax (log2 domain); scores for query nm live in-lane ----
    // s0[r] <-> key kb + r + 8*half ; s1[r] <-> key kb + 16 + r + 8*half
    if (MASKED) {
        const int lim = qb + nm - kb;           // key-in-tile <= lim is unmasked
#pragma unroll
        for (int r = 0; r < 8; ++r) {
            const int k0 = r + 8 * half;
            s0[r] = (k0      <= lim) ? s0[r] : -INFINITY;
            s1[r] = (k0 + 16 <= lim) ? s1[r] : -INFINITY;
        }
    }
    float tmax = -INFINITY;
#pragma unroll
    for (int r = 0; r < 8; ++r) tmax = fmaxf(tmax, fmaxf(s0[r], s1[r]));
    tmax = fmaxf(tmax, permx16f(tmax));
    const float mnew = fmaxf(m, tmax);
    float p0[8], p1[8], ls = 0.0f;
#pragma unroll
    for (int r = 0; r < 8; ++r) {
        p0[r] = exp2f(s0[r] - mnew);
        p1[r] = exp2f(s1[r] - mnew);
        ls += p0[r] + p1[r];
    }
    ls += permx16f(ls);
    const float alpha = exp2f(m - mnew);
    l = l * alpha + ls;
    m = mnew;
#pragma unroll
    for (int t = 0; t < 8; ++t) acc[t] *= alpha;

    // ---- P^T B-fragment (32 keys x 16 queries) straight from registers:
    // B dword j holds keys {2j,2j+1} of this half's 16-key range
    // (half0: keys 0..15, half1: keys 16..31); partner keys via permlanex16.
    v8u w;
#pragma unroll
    for (int j = 0; j < 4; ++j) {
        h2 lo = { (_Float16)p0[2 * j], (_Float16)p0[2 * j + 1] };
        h2 hi = { (_Float16)p1[2 * j], (_Float16)p1[2 * j + 1] };
        unsigned u0 = __builtin_bit_cast(unsigned, lo);
        unsigned u1 = __builtin_bit_cast(unsigned, hi);
        unsigned r0 = permx16u(u0);
        unsigned r1 = permx16u(u1);
        w[j]     = half ? r1 : u0;
        w[4 + j] = half ? u1 : r0;
    }
    v16h pb = __builtin_bit_cast(v16h, w);

    // ---- O^T += V^T_tile · P^T (Vt rows contiguous in s) ----
#pragma unroll
    for (int t = 0; t < 8; ++t) {
        const _Float16* vrow = vhead + (size_t)t * 16 * ATT_S + kb;
        v8h v0 = *(const v8h*)(vrow + half * 16);
        v8h v1 = *(const v8h*)(vrow + half * 16 + 8);
        v16h va = __builtin_shufflevector(v0, v1, 0,1,2,3,4,5,6,7,8,9,10,11,12,13,14,15);
        acc[t] = wmma_f16(va, pb, acc[t]);
    }
}

__global__ __launch_bounds__(32)
void fa_causal_f16(const _Float16* __restrict__ Qh,
                   const _Float16* __restrict__ Kh,
                   const _Float16* __restrict__ Vt,
                   float* __restrict__ O) {
    const int h    = blockIdx.y;
    const int qb   = blockIdx.x * 16;
    const int lane = threadIdx.x;
    const int nm   = lane & 15;
    const int half = lane >> 4;

    // Q as 4 B-fragments (32x16): lane N=query nm, elem i -> d = c*32+half*16+i
    v16h qf[4];
    {
        const _Float16* qrow = Qh + ((size_t)(qb + nm) * ATT_H + h) * ATT_D;
#pragma unroll
        for (int c = 0; c < 4; ++c)
            qf[c] = *(const v16h*)(qrow + c * 32 + half * 16);
    }

    v8f acc[8];
#pragma unroll
    for (int t = 0; t < 8; ++t) acc[t] = {};
    float m = -INFINITY, l = 0.0f;

    const _Float16* vhead = Vt + ((size_t)h * ATT_D + nm) * ATT_S;
    const int nblocks = (qb + 15) / 32 + 1;

    // main loop: blocks fully below the diagonal (kb+31 <= qb) -> no masking
    for (int b = 0; b < nblocks - 1; ++b)
        fa_block<false>(Kh, vhead, h, b * 32, qb, nm, half, qf, acc, m, l);
    // epilogue: diagonal block with causal mask
    fa_block<true>(Kh, vhead, h, (nblocks - 1) * 32, qb, nm, half, qf, acc, m, l);

    // normalize + store: lane writes query row qb+nm, d = t*16 + 8*half + [0..8)
    const float inv = 1.0f / l;
    float* orow = O + ((size_t)(qb + nm) * ATT_H + h) * ATT_D + 8 * half;
#pragma unroll
    for (int t = 0; t < 8; ++t) {
        v8f o = acc[t] * inv;
        *(float4*)(orow + t * 16)     = make_float4(o[0], o[1], o[2], o[3]);
        *(float4*)(orow + t * 16 + 4) = make_float4(o[4], o[5], o[6], o[7]);
    }
}

// =====================  fp32-direct fallback  =====================

__device__ __forceinline__ float groupmax16(float v) {
    v = fmaxf(v, __shfl_xor(v, 8, 32)); v = fmaxf(v, __shfl_xor(v, 4, 32));
    v = fmaxf(v, __shfl_xor(v, 2, 32)); v = fmaxf(v, __shfl_xor(v, 1, 32));
    return v;
}
__device__ __forceinline__ float groupsum16(float v) {
    v += __shfl_xor(v, 8, 32); v += __shfl_xor(v, 4, 32);
    v += __shfl_xor(v, 2, 32); v += __shfl_xor(v, 1, 32);
    return v;
}

__global__ __launch_bounds__(32)
void fa_causal_fp32(const float* __restrict__ Q, const float* __restrict__ K,
                    const float* __restrict__ V, float* __restrict__ O) {
    const int h = blockIdx.y, qb = blockIdx.x * 16;
    const int lane = threadIdx.x, nm = lane & 15, half = lane >> 4;
    __shared__ _Float16 Pl[16][40];
    v16h qa[4];
    {
        const float* qrow = Q + ((size_t)(qb + nm) * ATT_H + h) * ATT_D;
#pragma unroll
        for (int c = 0; c < 4; ++c) {
#pragma unroll
            for (int i = 0; i < 8; ++i) {
                qa[c][i]     = (_Float16)(qrow[c * 32 + half * 16 + i] * QK_SCALE);
                qa[c][8 + i] = (_Float16)(qrow[c * 32 + half * 16 + 8 + i] * QK_SCALE);
            }
        }
    }
    v8f acc[8];
#pragma unroll
    for (int t = 0; t < 8; ++t) acc[t] = {};
    float mrow[8], lrow[8];
#pragma unroll
    for (int r = 0; r < 8; ++r) { mrow[r] = -INFINITY; lrow[r] = 0.0f; }
    const int nblocks = (qb + 15) / 32 + 1;
    for (int b = 0; b < nblocks; ++b) {
        const int kb = b * 32;
        v8f s0 = {}, s1 = {};
        const float* kr0 = K + ((size_t)(kb + nm) * ATT_H + h) * ATT_D;
        const float* kr1 = K + ((size_t)(kb + 16 + nm) * ATT_H + h) * ATT_D;
#pragma unroll
        for (int c = 0; c < 4; ++c) {
            v16h kf0, kf1;
#pragma unroll
            for (int i = 0; i < 16; ++i) {
                kf0[i] = (_Float16)kr0[c * 32 + half * 16 + i];
                kf1[i] = (_Float16)kr1[c * 32 + half * 16 + i];
            }
            s0 = wmma_f16(qa[c], kf0, s0);
            s1 = wmma_f16(qa[c], kf1, s1);
        }
        __syncthreads();
#pragma unroll
        for (int r = 0; r < 8; ++r) {
            const int qi = qb + r + 8 * half;
            float a = (kb + nm <= qi) ? s0[r] : -INFINITY;
            float c = (kb + 16 + nm <= qi) ? s1[r] : -INFINITY;
            float mnew = fmaxf(mrow[r], groupmax16(fmaxf(a, c)));
            float p0 = __expf(a - mnew), p1 = __expf(c - mnew);
            float rs = groupsum16(p0 + p1);
            float alpha = __expf(mrow[r] - mnew);
            lrow[r] = lrow[r] * alpha + rs; mrow[r] = mnew;
#pragma unroll
            for (int t = 0; t < 8; ++t) acc[t][r] *= alpha;
            Pl[r + 8 * half][nm] = (_Float16)p0;
            Pl[r + 8 * half][nm + 16] = (_Float16)p1;
        }
        __syncthreads();
        v16h pa;
#pragma unroll
        for (int i = 0; i < 8; ++i) {
            pa[i]     = Pl[nm][half * 16 + i];
            pa[8 + i] = Pl[nm][half * 16 + 8 + i];
        }
        const float* vbase = V + ((size_t)(kb + half * 16) * ATT_H + h) * ATT_D + nm;
#pragma unroll
        for (int t = 0; t < 8; ++t) {
            v16h vf;
#pragma unroll
            for (int i = 0; i < 16; ++i)
                vf[i] = (_Float16)vbase[(size_t)i * ATT_H * ATT_D + t * 16];
            acc[t] = wmma_f16(pa, vf, acc[t]);
        }
    }
#pragma unroll
    for (int r = 0; r < 8; ++r) {
        float inv = 1.0f / lrow[r];
        float* orow = O + ((size_t)(qb + r + 8 * half) * ATT_H + h) * ATT_D;
#pragma unroll
        for (int t = 0; t < 8; ++t) orow[t * 16 + nm] = acc[t][r] * inv;
    }
}

// =====================  launch  =====================

extern "C" void kernel_launch(void* const* d_in, const int* in_sizes, int n_in,
                              void* d_out, int out_size, void* d_ws, size_t ws_size,
                              hipStream_t stream) {
    const float* Q = (const float*)d_in[0];
    const float* K = (const float*)d_in[1];
    const float* V = (const float*)d_in[2];
    float* O = (float*)d_out;

    const size_t n = (size_t)ATT_S * ATT_H * ATT_D;     // 4M elements
    const size_t need = 3 * n * sizeof(_Float16);       // 24 MB

    if (ws_size >= need) {
        _Float16* Qh = (_Float16*)d_ws;
        _Float16* Kh = Qh + n;
        _Float16* Vt = Kh + n;
        cvt_qk<<<dim3((unsigned)(n / 1024)), 256, 0, stream>>>(Q, K, Qh, Kh);
        cvt_v_t<<<dim3(ATT_S / 32, ATT_D / 32, ATT_H), 256, 0, stream>>>(V, Vt);
        fa_causal_f16<<<dim3(ATT_S / 16, ATT_H), 32, 0, stream>>>(Qh, Kh, Vt, O);
    } else {
        fa_causal_fp32<<<dim3(ATT_S / 16, ATT_H), 32, 0, stream>>>(Q, K, V, O);
    }
}
